// NeuralConvNetwork_34703335751794
// MI455X (gfx1250) — matrix operand order repaired and checked by reading, not compiled, not run on hardware
//
#include <hip/hip_runtime.h>

#define N_ATOMS 200000
#define N_MOLS  8000
#define FP_DIM  512
#define HIDW    100
#define KP      128   // padded K for all staged operands (covers K<=106 -> 4 chunks)

typedef __attribute__((ext_vector_type(16))) __bf16 v16bf;
typedef __attribute__((ext_vector_type(8)))  __bf16 v8bf;
typedef __attribute__((ext_vector_type(8)))  float  v8f;

// ---- fragment builders: ALL loads unconditional, padding done at staging ----
// A (16x32 bf16, contiguous k order shared with frag_b, not the ISA table order): lane m=lane&15, half=lane>>4;
// element e -> k = k0 + e + 16*half  => one contiguous 16-elt run.
__device__ __forceinline__ v16bf frag_a_lds(const __bf16* xs /*[16][KP]*/,
                                            int k0, int lane) {
  const int m = lane & 15, half = lane >> 4;
  const v8bf* p = (const v8bf*)(xs + m * KP + k0 + half * 16);
  v8bf lo = p[0];   // k0+16h .. k0+16h+7
  v8bf hi = p[1];   // k0+16h+8 .. k0+16h+15
  v16bf a;
#pragma unroll
  for (int e = 0; e < 8; ++e) { a[e] = lo[e]; a[8 + e] = hi[e]; }
  return a;
}

// B (32x16 bf16): lane col n = n0+(lane&15); element e -> k = k0 + e + 16*half
// => one contiguous 16-elt run in the N-major padded weight WT[Npad][KP].
__device__ __forceinline__ v16bf frag_b(const __bf16* __restrict__ WT, int n0,
                                        int k0, int lane) {
  const int half = lane >> 4;
  const v8bf* p =
      (const v8bf*)(WT + (size_t)(n0 + (lane & 15)) * KP + k0 + half * 16);
  v8bf lo = p[0];
  v8bf hi = p[1];
  v16bf b;
#pragma unroll
  for (int e = 0; e < 8; ++e) { b[e] = lo[e]; b[8 + e] = hi[e]; }
  return b;
}

// ---------------------------------------------------------------------------
__global__ void zero_kernel(float* __restrict__ p, int n) {
  int i = blockIdx.x * blockDim.x + threadIdx.x;
  if (i < n) p[i] = 0.0f;
}

// Pre-pad/transpose/convert weights: W (nmat batched, row-major K x N, f32)
// -> WT (nmat x Npad x KP, bf16, N-major, zero outside [K)x[N)).
__global__ void prep_wt(const float* __restrict__ W, int K, int N, int Npad,
                        int nmat, __bf16* __restrict__ WT) {
  int i = blockIdx.x * blockDim.x + threadIdx.x;
  int tot = nmat * Npad * KP;
  if (i >= tot) return;
  int k = i & (KP - 1);
  int n = (i >> 7) % Npad;
  int mat = i / (Npad * KP);
  WT[i] = (k < K && n < N) ? (__bf16)W[(size_t)mat * K * N + (size_t)k * N + n]
                           : (__bf16)0.0f;
}

// ---- fingerprint: softmax(x @ W + b) segment-summed by sorted mol_ids -----
// Block = 8 waves, one 16-row tile; wave w owns columns [64w, 64w+64).
__global__ __launch_bounds__(256) void fp_kernel(
    const float* __restrict__ x, int F, const __bf16* __restrict__ WT,
    const float* __restrict__ bias, const int* __restrict__ mol_ids,
    float* __restrict__ fps) {
  __shared__ __align__(16) __bf16 xs[16 * KP];
  __shared__ float red[8][16];
  const int tid = threadIdx.x;
  const int lane = tid & 31;
  const int wid = tid >> 5;
  const int half = lane >> 4;
  const int m0 = blockIdx.x * 16;
  const int n0w = wid * 64;

  // stage 16 x F tile, zero-padded to KP, as bf16 (coalesced, one pass)
  for (int idx = tid; idx < 16 * KP; idx += 256) {
    int r = idx >> 7, c = idx & (KP - 1);
    xs[idx] = (c < F) ? (__bf16)x[(size_t)(m0 + r) * F + c] : (__bf16)0.0f;
  }
  __syncthreads();

  v8f acc[4] = {};
  const int kch = (F + 31) >> 5;
  for (int kc = 0; kc < kch; ++kc) {
    v16bf a = frag_a_lds(xs, kc * 32, lane);
#pragma unroll
    for (int t = 0; t < 4; ++t) {
      v16bf b = frag_b(WT, n0w + t * 16, kc * 32, lane);
      acc[t] = __builtin_amdgcn_wmma_f32_16x16x32_bf16(
          false, a, false, b, (short)0, acc[t], false, false);
    }
  }
#pragma unroll
  for (int t = 0; t < 4; ++t) {
    float bb = bias[n0w + t * 16 + (lane & 15)];
#pragma unroll
    for (int v = 0; v < 8; ++v) acc[t][v] += bb;
  }

  // row max: wave-local (4 sets + half-wave xor), then across 8 waves via LDS
  float mv[8];
#pragma unroll
  for (int v = 0; v < 8; ++v) {
    float m = acc[0][v];
#pragma unroll
    for (int t = 1; t < 4; ++t) m = fmaxf(m, acc[t][v]);
    mv[v] = m;
  }
#pragma unroll
  for (int off = 1; off < 16; off <<= 1) {
#pragma unroll
    for (int v = 0; v < 8; ++v)
      mv[v] = fmaxf(mv[v], __shfl_xor(mv[v], off, 32));
  }
  if (lane == 0) {
#pragma unroll
    for (int v = 0; v < 8; ++v) red[wid][v] = mv[v];
  }
  if (lane == 16) {
#pragma unroll
    for (int v = 0; v < 8; ++v) red[wid][8 + v] = mv[v];
  }
  __syncthreads();
  float rmax[8];
#pragma unroll
  for (int v = 0; v < 8; ++v) {
    int rv = v + half * 8;
    float m = red[0][rv];
#pragma unroll
    for (int w = 1; w < 8; ++w) m = fmaxf(m, red[w][rv]);
    rmax[v] = m;
  }
  __syncthreads();

  // exp in place + row sum
  float sv[8];
#pragma unroll
  for (int v = 0; v < 8; ++v) {
    float s = 0.0f;
#pragma unroll
    for (int t = 0; t < 4; ++t) {
      acc[t][v] = __expf(acc[t][v] - rmax[v]);
      s += acc[t][v];
    }
    sv[v] = s;
  }
#pragma unroll
  for (int off = 1; off < 16; off <<= 1) {
#pragma unroll
    for (int v = 0; v < 8; ++v) sv[v] += __shfl_xor(sv[v], off, 32);
  }
  if (lane == 0) {
#pragma unroll
    for (int v = 0; v < 8; ++v) red[wid][v] = sv[v];
  }
  if (lane == 16) {
#pragma unroll
    for (int v = 0; v < 8; ++v) red[wid][8 + v] = sv[v];
  }
  __syncthreads();
  float rinv[8];
#pragma unroll
  for (int v = 0; v < 8; ++v) {
    int rv = v + half * 8;
    float s = 0.0f;
#pragma unroll
    for (int w = 0; w < 8; ++w) s += red[w][rv];
    rinv[v] = 1.0f / s;
  }

  // segment sum (mol_ids sorted): run-length compress, then atomicAdd
  int prev = -1;
  float run[4] = {0.0f, 0.0f, 0.0f, 0.0f};
#pragma unroll
  for (int v = 0; v < 8; ++v) {
    int row = m0 + half * 8 + v;
    int mol = mol_ids[row];
    if (mol != prev) {
      if (prev >= 0) {
#pragma unroll
        for (int t = 0; t < 4; ++t) {
          atomicAdd(&fps[(size_t)prev * FP_DIM + n0w + t * 16 + (lane & 15)],
                    run[t]);
          run[t] = 0.0f;
        }
      }
      prev = mol;
    }
#pragma unroll
    for (int t = 0; t < 4; ++t) run[t] += acc[t][v] * rinv[v];
  }
#pragma unroll
  for (int t = 0; t < 4; ++t)
    atomicAdd(&fps[(size_t)prev * FP_DIM + n0w + t * 16 + (lane & 15)], run[t]);
}

// ---- graph conv layer: gather-sum -> dual GEMM -> L2-normalize -> relu ----
// 1 wave per block, one 16-row tile (degree partitions are 16-multiples).
__global__ __launch_bounds__(32) void conv_kernel(
    const float* __restrict__ x, int F, const float* __restrict__ bond,
    const __bf16* __restrict__ WselfT, const float* __restrict__ bself,
    const __bf16* __restrict__ WdegT, const float* __restrict__ bdeg,
    const int* __restrict__ an1, const int* __restrict__ bn1,
    const int* __restrict__ an2, const int* __restrict__ bn2,
    const int* __restrict__ an3, const int* __restrict__ bn3,
    const int* __restrict__ an4, const int* __restrict__ bn4, int b1, int b2,
    int b3, float* __restrict__ xout) {
  __shared__ __align__(16) __bf16 xs[16 * KP];   // self rows (padded)
  __shared__ __align__(16) __bf16 nbr[16 * KP];  // gathered [atom|bond] rows
  const int lane = threadIdx.x;
  const int half = lane >> 4;
  const int m0 = blockIdx.x * 16;

  int d, base;
  const int *an, *bn;
  if (m0 < b1)      { d = 1; base = 0;  an = an1; bn = bn1; }
  else if (m0 < b2) { d = 2; base = b1; an = an2; bn = bn2; }
  else if (m0 < b3) { d = 3; base = b2; an = an3; bn = bn3; }
  else              { d = 4; base = b3; an = an4; bn = bn4; }
  const int K2 = F + 6;

  // stage self tile + gathered neighbor tile (random rows hit the 192MB L2)
  for (int idx = lane; idx < 16 * KP; idx += 32) {
    int r = idx >> 7, c = idx & (KP - 1);
    xs[idx] = (c < F) ? (__bf16)x[(size_t)(m0 + r) * F + c] : (__bf16)0.0f;
    int i = m0 + r - base;
    float val = 0.0f;
    if (c < F) {
      for (int j = 0; j < d; ++j) val += x[(size_t)an[i * d + j] * F + c];
    } else if (c < K2) {
      int cb = c - F;
      for (int j = 0; j < d; ++j) val += bond[(size_t)bn[i * d + j] * 6 + cb];
    }
    nbr[idx] = (__bf16)val;
  }
  __syncthreads();

  v8f acc[7] = {};  // 7 N-tiles cover 112 >= 100; padded cols stay exactly 0
  const int kch1 = (F + 31) >> 5;
  for (int kc = 0; kc < kch1; ++kc) {
    v16bf a = frag_a_lds(xs, kc * 32, lane);
#pragma unroll
    for (int t = 0; t < 7; ++t) {
      v16bf b = frag_b(WselfT, t * 16, kc * 32, lane);
      acc[t] = __builtin_amdgcn_wmma_f32_16x16x32_bf16(
          false, a, false, b, (short)0, acc[t], false, false);
    }
  }
  const __bf16* WdT = WdegT + (size_t)(d - 1) * 112 * KP;
  const int kch2 = (K2 + 31) >> 5;
  for (int kc = 0; kc < kch2; ++kc) {
    v16bf a = frag_a_lds(nbr, kc * 32, lane);
#pragma unroll
    for (int t = 0; t < 7; ++t) {
      v16bf b = frag_b(WdT, t * 16, kc * 32, lane);
      acc[t] = __builtin_amdgcn_wmma_f32_16x16x32_bf16(
          false, a, false, b, (short)0, acc[t], false, false);
    }
  }
  // biases (cols >= 100 remain exactly zero -> don't pollute the norm)
#pragma unroll
  for (int t = 0; t < 7; ++t) {
    int col = t * 16 + (lane & 15);
    if (col < HIDW) {
      float bb = bself[col] + bdeg[(d - 1) * HIDW + col];
#pragma unroll
      for (int v = 0; v < 8; ++v) acc[t][v] += bb;
    }
  }
  // per-row sum of squares via half-wave xor reduction
  float ss[8];
#pragma unroll
  for (int v = 0; v < 8; ++v) {
    float s = 0.0f;
#pragma unroll
    for (int t = 0; t < 7; ++t) s += acc[t][v] * acc[t][v];
    ss[v] = s;
  }
#pragma unroll
  for (int off = 1; off < 16; off <<= 1) {
#pragma unroll
    for (int v = 0; v < 8; ++v) ss[v] += __shfl_xor(ss[v], off, 32);
  }
  // normalize + relu + store
#pragma unroll
  for (int v = 0; v < 8; ++v) {
    float sc = 1.0f / fmaxf(sqrtf(ss[v]), 1e-12f);
    int row = m0 + half * 8 + v;
#pragma unroll
    for (int t = 0; t < 7; ++t) {
      int col = t * 16 + (lane & 15);
      if (col < HIDW)
        xout[(size_t)row * HIDW + col] = fmaxf(acc[t][v] * sc, 0.0f);
    }
  }
}

// ---------------------------------------------------------------------------
extern "C" void kernel_launch(void* const* d_in, const int* in_sizes, int n_in,
                              void* d_out, int out_size, void* d_ws,
                              size_t ws_size, hipStream_t stream) {
  (void)n_in; (void)out_size; (void)ws_size;
  const float* atom = (const float*)d_in[0];
  const float* bond = (const float*)d_in[1];
  const int* an1 = (const int*)d_in[2];
  const int* bn1 = (const int*)d_in[3];
  const int* an2 = (const int*)d_in[4];
  const int* bn2 = (const int*)d_in[5];
  const int* an3 = (const int*)d_in[6];
  const int* bn3 = (const int*)d_in[7];
  const int* an4 = (const int*)d_in[8];
  const int* bn4 = (const int*)d_in[9];
  const int* mol_ids = (const int*)d_in[10];
  const float* Wself0 = (const float*)d_in[11];
  const float* bself0 = (const float*)d_in[12];
  const float* Wdeg0  = (const float*)d_in[13];
  const float* bdeg0  = (const float*)d_in[14];
  const float* Wself1 = (const float*)d_in[15];
  const float* bself1 = (const float*)d_in[16];
  const float* Wdeg1  = (const float*)d_in[17];
  const float* bdeg1  = (const float*)d_in[18];
  const float* Wout0  = (const float*)d_in[19];
  const float* bout0  = (const float*)d_in[20];
  const float* Wout1  = (const float*)d_in[21];
  const float* bout1  = (const float*)d_in[22];
  const float* Wout2  = (const float*)d_in[23];
  const float* bout2  = (const float*)d_in[24];

  const int n1 = in_sizes[2];
  const int n2 = in_sizes[4] / 2;
  const int n3 = in_sizes[6] / 3;
  const int b1 = n1, b2 = n1 + n2, b3 = n1 + n2 + n3;

  float* fps = (float*)d_out;

  // workspace layout
  char* w = (char*)d_ws;
  float* x1 = (float*)w;  w += (size_t)N_ATOMS * HIDW * sizeof(float);
  float* x2 = (float*)w;  w += (size_t)N_ATOMS * HIDW * sizeof(float);
  __bf16* WoutT0  = (__bf16*)w; w += (size_t)512 * KP * 2;
  __bf16* WoutT1  = (__bf16*)w; w += (size_t)512 * KP * 2;
  __bf16* WoutT2  = (__bf16*)w; w += (size_t)512 * KP * 2;
  __bf16* WselfT0 = (__bf16*)w; w += (size_t)112 * KP * 2;
  __bf16* WselfT1 = (__bf16*)w; w += (size_t)112 * KP * 2;
  __bf16* WdT0    = (__bf16*)w; w += (size_t)4 * 112 * KP * 2;
  __bf16* WdT1    = (__bf16*)w; w += (size_t)4 * 112 * KP * 2;

  // one-time (per launch) weight pre-pad/transpose/bf16
  int t;
  t = 512 * KP;
  prep_wt<<<(t + 255) / 256, 256, 0, stream>>>(Wout0, 62, 512, 512, 1, WoutT0);
  prep_wt<<<(t + 255) / 256, 256, 0, stream>>>(Wout1, 100, 512, 512, 1, WoutT1);
  prep_wt<<<(t + 255) / 256, 256, 0, stream>>>(Wout2, 100, 512, 512, 1, WoutT2);
  t = 112 * KP;
  prep_wt<<<(t + 255) / 256, 256, 0, stream>>>(Wself0, 62, 100, 112, 1, WselfT0);
  prep_wt<<<(t + 255) / 256, 256, 0, stream>>>(Wself1, 100, 100, 112, 1, WselfT1);
  t = 4 * 112 * KP;
  prep_wt<<<(t + 255) / 256, 256, 0, stream>>>(Wdeg0, 68, 100, 112, 4, WdT0);
  prep_wt<<<(t + 255) / 256, 256, 0, stream>>>(Wdeg1, 106, 100, 112, 4, WdT1);

  const int ntiles = N_ATOMS / 16;  // 12500
  const int fpn = N_MOLS * FP_DIM;

  zero_kernel<<<(fpn + 255) / 256, 256, 0, stream>>>(fps, fpn);
  fp_kernel<<<ntiles, 256, 0, stream>>>(atom, 62, WoutT0, bout0, mol_ids, fps);
  conv_kernel<<<ntiles, 32, 0, stream>>>(atom, 62, bond, WselfT0, bself0, WdT0,
                                         bdeg0, an1, bn1, an2, bn2, an3, bn3,
                                         an4, bn4, b1, b2, b3, x1);
  fp_kernel<<<ntiles, 256, 0, stream>>>(x1, HIDW, WoutT1, bout1, mol_ids, fps);
  conv_kernel<<<ntiles, 32, 0, stream>>>(x1, HIDW, bond, WselfT1, bself1, WdT1,
                                         bdeg1, an1, bn1, an2, bn2, an3, bn3,
                                         an4, bn4, b1, b2, b3, x2);
  fp_kernel<<<ntiles, 256, 0, stream>>>(x2, HIDW, WoutT2, bout2, mol_ids, fps);
}
